// RINet_attention_71949292142744
// MI455X (gfx1250) — matrix operand order repaired and checked by reading, not compiled, not run on hardware
//
#include <hip/hip_runtime.h>
#include <math.h>

// ---------------------------------------------------------------------------
// RINet on MI455X (gfx1250).  All convs + scoring GEMM on v_wmma_f32_16x16x32_f16.
// im2col staged in LDS as f16 (A-frags = 2x ds_load_b128); weights pre-packed
// into per-lane B-fragment layout in global (L2-resident).
// ---------------------------------------------------------------------------

typedef __attribute__((ext_vector_type(16))) _Float16 v16h;
typedef __attribute__((ext_vector_type(8)))  _Float16 v8h;
typedef __attribute__((ext_vector_type(2)))  _Float16 v2h;
typedef __attribute__((ext_vector_type(8)))  float    v8f;

#define NB 2048
#define NT 9
#define BLK 256            // feature kernel: 8 wave32
#define SBLK 128           // score kernel: 4 wave32

// workspace layout (float units)
#define WS_FEAT   0
#define WS_DIFF   (NT * NB * 288)            // 5308416
#define WS_BN     (WS_DIFF + 8 * NB)         // 5324800, 600 floats (scale,shift)
#define WS_HALF   (WS_BN + 600)              // halves region (16B aligned)
#define BP_HALVES 38912                      // packed conv weights

struct FeatParams {
  const float* data[9];
  const float* attW[8]; const float* attb[8];
  const float* c1w[6];  const float* c1b[6];
};

struct PrepParams {
  const float* cw[7]; const float* cb[7]; const float* cg[7];
  const float* cbeta[7]; const float* cmu[7]; const float* cvar[7];
  const float* W1;
};

// ------- channel attention: x *= sigmoid(mean_L(x) @ W^T + b) ---------------
// Grouped reduction: G = BLK/Cp threads per channel (Cp = pow2 >= C).
__device__ __forceinline__ void attention(float* x, int C, int L, int logCp,
                                          const float* W, const float* bias,
                                          float* mean, float* red, float* gate,
                                          int tid) {
  const int Cp = 1 << logCp, G = BLK >> logCp;
  const int c = tid & (Cp - 1), g = tid >> logCp;
  float s = 0.f;
  if (c < C) {
    const float* row = x + c * L;
    for (int l = g; l < L; l += G) s += row[l];
  }
  red[tid] = s;
  __syncthreads();
  if (tid < C) {
    float m = 0.f;
    for (int k = 0; k < G; ++k) m += red[tid + (k << logCp)];
    mean[tid] = m / (float)L;
  }
  __syncthreads();
  if (tid < C) {
    float d = bias[tid];
    const float* wr = W + tid * C;
    for (int cc = 0; cc < C; ++cc) d += mean[cc] * wr[cc];
    gate[tid] = 1.f / (1.f + __expf(-d));
  }
  __syncthreads();
  for (int i = tid; i < C * L; i += BLK) x[i] *= gate[i / L];
  __syncthreads();
}

// ------- build f16 im2col in LDS: imc[m*W + (ci*3+dk)] ----------------------
__device__ __forceinline__ void im2col(const float* x, _Float16* imc,
                                       int Cin, int L, int W, int Lpad, int tid) {
  const int wid = tid >> 5, lane = tid & 31;
  const int K3 = 3 * Cin;
  for (int K = wid; K < W; K += 8) {
    const bool real = (K < K3);
    int ci = 0, dk = 0;
    if (real) { ci = K / 3; dk = K - 3 * ci; }
    const float* row = x + ci * L;
    for (int m = lane; m < Lpad; m += 32) {
      float v = 0.f;
      if (real) {
        int l = m + dk;
        while (l >= L) l -= L;
        v = row[l];
      }
      imc[m * W + K] = (_Float16)v;
    }
  }
  __syncthreads();
}

// ------- WMMA conv GEMM + fused BN + LeakyReLU, writes in place -------------
__device__ __forceinline__ void conv_gemm(const _Float16* imc, float* yout,
                                          int Cout, int L, int W, int Lpad,
                                          int nK, int lognN,
                                          const _Float16* bw, const float* bn,
                                          int tid) {
  const int wid = tid >> 5, lane = tid & 31;
  const int nl = lane & 15;
  const int koffA = (lane >> 4) * 8;
  const int nN = 1 << lognN;
  const int jobs = (Lpad >> 4) << lognN;
  for (int job = wid; job < jobs; job += 8) {
    const int mt = job >> lognN, ntile = job & (nN - 1);
    const int n = (ntile << 4) + nl;
    const _Float16* arow = imc + (mt * 16 + nl) * W;
    const _Float16* bfr  = bw + ((ntile * nK) << 9) + lane * 16;
    v8f acc = {};
    for (int kt = 0; kt < nK; ++kt) {
      const int st = kt * 32 + koffA;
      v8h z = {};
      union { v16h v; v8h h[2]; } au;
      au.h[0] = (st      < W) ? *(const v8h*)(arow + st)      : z;
      au.h[1] = (st + 16 < W) ? *(const v8h*)(arow + st + 16) : z;
      const v16h bv = *(const v16h*)(bfr + (kt << 9));
      acc = __builtin_amdgcn_wmma_f32_16x16x32_f16(false, au.v, false, bv,
                                                   (short)0, acc, false, false);
    }
    if (n < Cout) {
      const float scale = bn[2 * n], shift = bn[2 * n + 1];
      const int mbase = mt * 16 + ((lane >> 4) << 3);
      float yv[8];
      #pragma unroll
      for (int r = 0; r < 8; ++r) {
        const float v = acc[r] * scale + shift;
        yv[r] = v >= 0.f ? v : 0.1f * v;
      }
      float* dst = yout + n * L + mbase;
      if (mbase + 8 <= L) {
        #pragma unroll
        for (int r = 0; r < 8; ++r) dst[r] = yv[r];    // mergeable stores
      } else {
        #pragma unroll
        for (int r = 0; r < 8; ++r) if (mbase + r < L) dst[r] = yv[r];
      }
    }
  }
  __syncthreads();
}

// ------- pick stride-r phase with largest L1 mass, compact IN PLACE ---------
__device__ __forceinline__ void down(float* x, int C, int L, int r,
                                     float* red, int* bestp, int tid) {
  const int Ld = L / r;
  for (int p = 0; p < r; ++p) {
    float s = 0.f;
    for (int i = tid; i < C * Ld; i += BLK) {
      int c = i / Ld, l = i - c * Ld;
      s += fabsf(x[c * L + l * r + p]);
    }
    red[p * BLK + tid] = s;
  }
  __syncthreads();
  for (int off = BLK / 2; off > 0; off >>= 1) {
    if (tid < off)
      for (int p = 0; p < r; ++p) red[p * BLK + tid] += red[p * BLK + tid + off];
    __syncthreads();
  }
  if (tid == 0) {
    float best = red[0]; int bi = 0;
    for (int p = 1; p < r; ++p)
      if (red[p * BLK] > best) { best = red[p * BLK]; bi = p; }
    *bestp = bi;
  }
  __syncthreads();
  const int bp = *bestp;
  float tmp[8];
  #pragma unroll
  for (int u = 0; u < 8; ++u) {
    int i = tid + u * BLK;
    if (i < C * Ld) { int c = i / Ld, l = i - c * Ld; tmp[u] = x[c * L + l * r + bp]; }
  }
  __syncthreads();
  #pragma unroll
  for (int u = 0; u < 8; ++u) {
    int i = tid + u * BLK;
    if (i < C * Ld) x[i] = tmp[u];
  }
  __syncthreads();
}

// ------- 1x1 head, grouped reduction like attention -------------------------
__device__ __forceinline__ void c1x1(const float* x, int C, int L, int logCp,
                                     const float* w, const float* b,
                                     float* red, float* fout, int tid) {
  const int Cp = 1 << logCp, G = BLK >> logCp;
  const int c = tid & (Cp - 1), g = tid >> logCp;
  float s = 0.f;
  if (c < C) {
    const float* row = x + c * L;
    for (int l = g; l < L; l += G) s += row[l] * w[l];
  }
  red[tid] = s;
  __syncthreads();
  if (tid < C) {
    float acc = b[0];
    for (int k = 0; k < G; ++k) acc += red[tid + (k << logCp)];
    fout[tid] = acc;
  }
  __syncthreads();
}

// ---------------------------------------------------------------------------
__global__ __launch_bounds__(BLK) void feature_kernel(FeatParams P,
                                                      float* feat_g,
                                                      const float* bn,
                                                      const _Float16* bp) {
  __shared__ __align__(16) float bufA[16 * 360];
  __shared__ __align__(16) _Float16 imc[368 * 40];
  __shared__ float mean[128];
  __shared__ float gate[128];
  __shared__ float red[3 * BLK];
  __shared__ __align__(16) float feat[288];
  __shared__ int   bestp;

  const int tid = threadIdx.x;
  const int s = blockIdx.x;
  const int t = s / NB, b = s - t * NB;
  const float4* xin4 = (const float4*)(P.data[t] + (size_t)b * 12 * 360);

  for (int i = tid; i < (12 * 360) / 4; i += BLK) ((float4*)bufA)[i] = xin4[i];
  __syncthreads();

  // layer1a: 12->12, L=360
  attention(bufA, 12, 360, 4, P.attW[0], P.attb[0], mean, red, gate, tid);
  im2col(bufA, imc, 12, 360, 40, 368, tid);
  conv_gemm(imc, bufA, 12, 360, 40, 368, 2, 0, bp + 0, bn + 0, tid);
  // layer1b: 12->16
  attention(bufA, 12, 360, 4, P.attW[1], P.attb[1], mean, red, gate, tid);
  im2col(bufA, imc, 12, 360, 40, 368, tid);
  conv_gemm(imc, bufA, 16, 360, 40, 368, 2, 0, bp + 1024, bn + 24, tid);
  attention(bufA, 16, 360, 4, P.attW[2], P.attb[2], mean, red, gate, tid);
  c1x1(bufA, 16, 360, 4, P.c1w[0], P.c1b[0], red, feat + 0, tid);

  down(bufA, 16, 360, 3, red, &bestp, tid);
  im2col(bufA, imc, 16, 120, 48, 128, tid);
  conv_gemm(imc, bufA, 16, 120, 48, 128, 2, 0, bp + 2048, bn + 56, tid);
  attention(bufA, 16, 120, 4, P.attW[3], P.attb[3], mean, red, gate, tid);
  c1x1(bufA, 16, 120, 4, P.c1w[1], P.c1b[1], red, feat + 16, tid);

  down(bufA, 16, 120, 3, red, &bestp, tid);
  im2col(bufA, imc, 16, 40, 48, 48, tid);
  conv_gemm(imc, bufA, 32, 40, 48, 48, 2, 1, bp + 3072, bn + 88, tid);
  attention(bufA, 32, 40, 5, P.attW[4], P.attb[4], mean, red, gate, tid);
  c1x1(bufA, 32, 40, 5, P.c1w[2], P.c1b[2], red, feat + 32, tid);

  down(bufA, 32, 40, 2, red, &bestp, tid);
  im2col(bufA, imc, 32, 20, 96, 32, tid);
  conv_gemm(imc, bufA, 32, 20, 96, 32, 3, 1, bp + 5120, bn + 152, tid);
  attention(bufA, 32, 20, 5, P.attW[5], P.attb[5], mean, red, gate, tid);
  c1x1(bufA, 32, 20, 5, P.c1w[3], P.c1b[3], red, feat + 64, tid);

  down(bufA, 32, 20, 2, red, &bestp, tid);
  im2col(bufA, imc, 32, 10, 96, 16, tid);
  conv_gemm(imc, bufA, 64, 10, 96, 16, 3, 2, bp + 8192, bn + 216, tid);
  attention(bufA, 64, 10, 6, P.attW[6], P.attb[6], mean, red, gate, tid);
  c1x1(bufA, 64, 10, 6, P.c1w[4], P.c1b[4], red, feat + 96, tid);

  down(bufA, 64, 10, 2, red, &bestp, tid);
  im2col(bufA, imc, 64, 5, 192, 16, tid);
  conv_gemm(imc, bufA, 128, 5, 192, 16, 6, 3, bp + 14336, bn + 344, tid);
  attention(bufA, 128, 5, 7, P.attW[7], P.attb[7], mean, red, gate, tid);
  c1x1(bufA, 128, 5, 7, P.c1w[5], P.c1b[5], red, feat + 160, tid);

  float4* fo = (float4*)(feat_g + (size_t)s * 288);
  for (int i = tid; i < 288 / 4; i += BLK) fo[i] = ((float4*)feat)[i];
}

// ---------------------------------------------------------------------------
// Prep: pack conv weights + lin1W into per-lane B-fragment layout (f16),
// fold BN into (scale, shift).
// ---------------------------------------------------------------------------
__global__ __launch_bounds__(256) void prep_kernel(PrepParams P, float* bn,
                                                   _Float16* bp, _Float16* w1p) {
  const int gsz = gridDim.x * blockDim.x;
  const int gid = blockIdx.x * blockDim.x + threadIdx.x;
  const int Cin[7]   = {12, 12, 16, 16, 32, 32, 64};
  const int Cout[7]  = {12, 16, 16, 32, 32, 64, 128};
  const int nKl[7]   = {2, 2, 2, 2, 3, 3, 6};
  const int nNl[7]   = {1, 1, 1, 2, 2, 4, 8};
  const int bwoff[7] = {0, 1024, 2048, 3072, 5120, 8192, 14336};
  const int bnoff[7] = {0, 24, 56, 88, 152, 216, 344};
  for (int layer = 0; layer < 7; ++layer) {
    const int K3 = 3 * Cin[layer];
    const int nE = nNl[layer] * nKl[layer] * 512;
    const float* w = P.cw[layer];
    for (int i = gid; i < nE; i += gsz) {
      const int j = i & 15, lane = (i >> 4) & 31, frag = i >> 9;
      const int kt = frag % nKl[layer], ntile = frag / nKl[layer];
      const int n = ntile * 16 + (lane & 15);
      const int K = kt * 32 + (lane >> 4) * 8 + (j & 7) + (j >> 3) * 16;
      float v = 0.f;
      if (n < Cout[layer] && K < K3) {
        const int ci = K / 3, dk = K - 3 * ci;
        v = w[(n * Cin[layer] + ci) * 3 + dk];
      }
      bp[bwoff[layer] + i] = (_Float16)v;
    }
    for (int n = gid; n < Cout[layer]; n += gsz) {
      const float sc = P.cg[layer][n] * rsqrtf(P.cvar[layer][n] + 1e-5f);
      bn[bnoff[layer] + 2 * n]     = sc;
      bn[bnoff[layer] + 2 * n + 1] = (P.cb[layer][n] - P.cmu[layer][n]) * sc
                                     + P.cbeta[layer][n];
    }
  }
  for (int i = gid; i < 8 * 9 * 512; i += gsz) {
    const int j = i & 15, lane = (i >> 4) & 31, frag = i >> 9;
    const int kt = frag % 9, ntile = frag / 9;
    const int n = ntile * 16 + (lane & 15);
    const int K = kt * 32 + (lane >> 4) * 8 + (j & 7) + (j >> 3) * 16;
    w1p[i] = (_Float16)P.W1[n * 288 + K];
  }
}

// ---------------------------------------------------------------------------
// Scoring: block stages d=|fx-fy| (64 rows x 288) into LDS f16, then 4 waves
// run 16x128x288 WMMA strips; fused bias+lrelu+dot(lin2)+sigmoid epilogue.
// ---------------------------------------------------------------------------
__global__ __launch_bounds__(SBLK) void score_kernel(const float* __restrict__ feat_g,
                                                     const _Float16* __restrict__ w1p,
                                                     const float* __restrict__ B1,
                                                     const float* __restrict__ W2,
                                                     const float* __restrict__ B2,
                                                     float* __restrict__ out_cat,
                                                     float* __restrict__ diff_cat) {
  __shared__ __align__(16) _Float16 dsh[64 * 288];
  __shared__ float dsrow[64];

  const int tid = threadIdx.x, wid = tid >> 5, lane = tid & 31;
  const int nl = lane & 15;
  const int m0 = blockIdx.x * 64;          // rows m0..m0+63, all same t
  const int t = m0 >> 11;

  // stage |fx - fy| (float2 loads, packed f16 stores) + per-row sum-sq
  for (int rr = 0; rr < 16; ++rr) {
    const int row = wid * 16 + rr;
    const int b = (m0 + row) & (NB - 1);
    const float* px = feat_g + (size_t)b * 288;
    const float* py = feat_g + ((size_t)(t + 1) * NB + b) * 288;
    float s2 = 0.f;
    #pragma unroll
    for (int u = 0; u < 5; ++u) {
      const int K2 = u * 64 + lane * 2;
      if (K2 < 288) {
        const float2 vx = *(const float2*)(px + K2);
        const float2 vy = *(const float2*)(py + K2);
        const float d0 = fabsf(vx.x - vy.x);
        const float d1 = fabsf(vx.y - vy.y);
        s2 += d0 * d0 + d1 * d1;
        v2h h; h.x = (_Float16)d0; h.y = (_Float16)d1;
        *(v2h*)(dsh + row * 288 + K2) = h;
      }
    }
    #pragma unroll
    for (int mask = 16; mask > 0; mask >>= 1) s2 += __shfl_xor(s2, mask);
    if (lane == 0) dsrow[row] = s2;
  }
  __syncthreads();

  // WMMA: one 16-row M tile per wave, K=288 (9 tiles), N=128 (8 tiles)
  const int koffA = (lane >> 4) * 8;
  const _Float16* arow = dsh + (wid * 16 + nl) * 288;
  v8f acc[8] = {};
  for (int kt = 0; kt < 9; ++kt) {
    union { v16h v; v8h h[2]; } au;
    au.h[0] = *(const v8h*)(arow + kt * 32 + koffA);
    au.h[1] = *(const v8h*)(arow + kt * 32 + koffA + 16);
    #pragma unroll
    for (int nt = 0; nt < 8; ++nt) {
      const v16h bv = *(const v16h*)(w1p + (((nt * 9 + kt) << 5) + lane) * 16);
      acc[nt] = __builtin_amdgcn_wmma_f32_16x16x32_f16(false, au.v, false, bv,
                                                       (short)0, acc[nt], false, false);
    }
  }

  // epilogue: h = lrelu(acc + b1); s = sigmoid(h.w2 + b2)
  float pr[8];
  #pragma unroll
  for (int r = 0; r < 8; ++r) pr[r] = 0.f;
  #pragma unroll
  for (int nt = 0; nt < 8; ++nt) {
    const int n = nt * 16 + nl;
    const float wv = W2[n], bv = B1[n];
    #pragma unroll
    for (int r = 0; r < 8; ++r) {
      float h = acc[nt][r] + bv;
      h = h >= 0.f ? h : 0.1f * h;
      pr[r] += h * wv;
    }
  }
  #pragma unroll
  for (int mask = 1; mask < 16; mask <<= 1)
    #pragma unroll
    for (int r = 0; r < 8; ++r) pr[r] += __shfl_xor(pr[r], mask);

  const float b2s = B2[0];
  if (nl == 0) {
    const int half = lane >> 4;
    #pragma unroll
    for (int r = 0; r < 8; ++r) {
      const int m = m0 + wid * 16 + half * 8 + r;
      out_cat[m] = 1.f / (1.f + __expf(-(pr[r] + b2s)));
    }
  }
  if (tid < 64) diff_cat[m0 + tid] = sqrtf(dsrow[tid]);
}

// ---------------------------------------------------------------------------
__global__ __launch_bounds__(256) void finalize_kernel(const float* __restrict__ out_cat,
                                                       const float* __restrict__ diff_cat,
                                                       float* __restrict__ out,
                                                       float* __restrict__ diff) {
  const int b = blockIdx.x * blockDim.x + threadIdx.x;
  if (b < NB) {
    float best = out_cat[b];
    int bi = 0;
    for (int t = 1; t < 8; ++t) {
      float v = out_cat[t * NB + b];
      if (v > best) { best = v; bi = t; }
    }
    out[b] = best;
    diff[b] = diff_cat[bi * NB + b];
  }
}

// ---------------------------------------------------------------------------
extern "C" void kernel_launch(void* const* d_in, const int* in_sizes, int n_in,
                              void* d_out, int out_size, void* d_ws, size_t ws_size,
                              hipStream_t stream) {
  (void)in_sizes; (void)n_in; (void)out_size; (void)ws_size;
  const float* const* in = (const float* const*)d_in;

  // jax pytree flatten order (sorted dict keys)
  FeatParams FP;
  PrepParams PP;
  for (int i = 0; i < 9; ++i) FP.data[i] = in[i];
  for (int i = 0; i < 8; ++i) { FP.attW[i] = in[9 + 2 * i]; FP.attb[i] = in[10 + 2 * i]; }
  for (int i = 0; i < 6; ++i) { FP.c1b[i] = in[25 + 2 * i]; FP.c1w[i] = in[26 + 2 * i]; }
  for (int i = 0; i < 7; ++i) {
    const int base = 37 + 6 * i;
    PP.cb[i]    = in[base + 0];
    PP.cbeta[i] = in[base + 1];
    PP.cg[i]    = in[base + 2];
    PP.cmu[i]   = in[base + 3];
    PP.cvar[i]  = in[base + 4];
    PP.cw[i]    = in[base + 5];
  }
  PP.W1 = in[79];
  const float* B1 = in[80];
  const float* W2 = in[81];
  const float* B2 = in[82];

  float* wsf      = (float*)d_ws;
  float* feat_g   = wsf + WS_FEAT;
  float* diff_cat = wsf + WS_DIFF;
  float* bn       = wsf + WS_BN;
  _Float16* bp    = (_Float16*)(wsf + WS_HALF);
  _Float16* w1p   = bp + BP_HALVES;

  float* out     = (float*)d_out;
  float* diffp   = out + NB;
  float* out_cat = out + 2 * NB;

  prep_kernel<<<64, 256, 0, stream>>>(PP, bn, bp, w1p);
  feature_kernel<<<NT * NB, BLK, 0, stream>>>(FP, feat_g, bn, bp);
  score_kernel<<<(8 * NB) / 64, SBLK, 0, stream>>>(feat_g, w1p, B1, W2, B2,
                                                   out_cat, diff_cat);
  finalize_kernel<<<NB / 256, 256, 0, stream>>>(out_cat, diff_cat, out, diffp);
}
